// Triton_34789235097626
// MI455X (gfx1250) — matrix operand order repaired and checked by reading, not compiled, not run on hardware
//
#include <hip/hip_runtime.h>

// ---------------------------------------------------------------------------
// MaskedLinear on MI455X (gfx1250):
//   out[tok] = x[tok] @ (mask[tok] ? Wv : Wt)^T
//
// Strategy:
//  1) Compact token indices into two lists (visual / text): halves FLOPs vs
//     the reference, which computes both GEMMs and selects.
//  2) Gather-GEMM per group using v_wmma_f32_16x16x32_bf16 (f32 accumulate),
//     128x128x32 tiles, double-buffered LDS, wave32 / 8 waves per block.
//  3) Packed f32->bf16 conversion (v_cvt_pk_bf16_f32) during LDS staging;
//     non-temporal C stores to keep x/W resident in the 192 MB L2.
// ---------------------------------------------------------------------------

typedef __attribute__((ext_vector_type(16))) __bf16 v16bf;
typedef __attribute__((ext_vector_type(8)))  __bf16 v8bf;
typedef __attribute__((ext_vector_type(2)))  __bf16 v2bf;
typedef __attribute__((ext_vector_type(2)))  float  v2f;
typedef __attribute__((ext_vector_type(8)))  float  v8f;

#define DK     4096
#define DN     4096
#define MTOK   16384
#define TILE_M 128
#define TILE_N 128
#define TILE_K 32
#define NKSTEP (DK / TILE_K)

__global__ void zero_counters(int* c) {
  if (threadIdx.x < 2) c[threadIdx.x] = 0;
}

// Wave-aggregated compaction: one ballot + one atomicAdd per wave per list.
__global__ void compact_mask(const unsigned char* __restrict__ mask,
                             int* __restrict__ cnts,
                             int* __restrict__ idxV,
                             int* __restrict__ idxT) {
  const int i    = blockIdx.x * 256 + threadIdx.x;   // MTOK % 256 == 0
  const int lane = threadIdx.x & 31;
  const bool isV = (mask[i] != 0);

  const unsigned mv = (unsigned)__ballot(isV);
  const int nv = __popc(mv);

  int baseV = 0, baseT = 0;
  if (lane == 0) {
    baseV = atomicAdd(&cnts[0], nv);
    baseT = atomicAdd(&cnts[1], 32 - nv);
  }
  baseV = __shfl(baseV, 0);
  baseT = __shfl(baseT, 0);

  const unsigned below = mv & ((1u << lane) - 1u);
  const int rv = __popc(below);          // rank among V lanes below me
  if (isV) idxV[baseV + rv]          = i;
  else     idxT[baseT + (lane - rv)] = i;
}

// Packed f32 pair -> bf16 pair (should lower to v_cvt_pk_bf16_f32).
__device__ __forceinline__ unsigned pkbf(float a, float b) {
  v2f s; s.x = a; s.y = b;
  v2bf h = __builtin_convertvector(s, v2bf);
  union { v2bf h; unsigned u; } c;
  c.h = h;
  return c.u;
}

__device__ __forceinline__ void gload16(const float* __restrict__ p, float* f) {
  const float4* s = (const float4*)p;
  *(float4*)&f[0]  = s[0];
  *(float4*)&f[4]  = s[1];
  *(float4*)&f[8]  = s[2];
  *(float4*)&f[12] = s[3];
}

__device__ __forceinline__ void lstore16(unsigned short* dst, const float* f) {
  uint4* d = (uint4*)dst;
  d[0] = make_uint4(pkbf(f[0], f[1]),  pkbf(f[2], f[3]),
                    pkbf(f[4], f[5]),  pkbf(f[6], f[7]));
  d[1] = make_uint4(pkbf(f[8], f[9]),  pkbf(f[10], f[11]),
                    pkbf(f[12], f[13]), pkbf(f[14], f[15]));
}

// Build a 16-element bf16 fragment from two 16-byte LDS chunks.
__device__ __forceinline__ v16bf frag(const unsigned short* p0,
                                      const unsigned short* p1) {
  v8bf a = *(const v8bf*)p0;
  v8bf b = *(const v8bf*)p1;
  return __builtin_shufflevector(a, b, 0, 1, 2, 3, 4, 5, 6, 7,
                                        8, 9, 10, 11, 12, 13, 14, 15);
}

__global__ __launch_bounds__(256)
void masked_gemm(const float* __restrict__ x,
                 const float* __restrict__ Wv,
                 const float* __restrict__ Wt,
                 float* __restrict__ out,
                 const int* __restrict__ cnts,
                 const int* __restrict__ idxV,
                 const int* __restrict__ idxT) {
  const int grp   = blockIdx.z;
  const int count = cnts[grp];
  const int m0    = blockIdx.y * TILE_M;
  if (m0 >= count) return;                 // uniform exit, no barriers yet
  const int n0 = blockIdx.x * TILE_N;
  const int* __restrict__ idx = grp ? idxT : idxV;
  const float* __restrict__ W = grp ? Wt : Wv;

  __shared__ __align__(16) unsigned short lA[2][TILE_M * TILE_K];  // 8 KB x2
  __shared__ __align__(16) unsigned short lB[2][TILE_N * TILE_K];  // 8 KB x2

  const int tid  = threadIdx.x;
  // Staging: each thread owns half a 32-element row (16 f32 = 64 B).
  const int r    = tid >> 1;               // 0..127 tile row / W row
  const int half = tid & 1;                // k sub-offset 0 / 16
  const int gm   = m0 + r;
  const int tokA = (gm < count) ? idx[gm] : 0;      // pad rows read token 0
  const float* aSrc = x + (size_t)tokA * DK + half * 16;
  const float* bSrc = W + (size_t)(n0 + r) * DK + half * 16;
  const int sOff = r * TILE_K + half * 16;

  // Compute mapping: 8 waves in a 2(M) x 4(N) grid; wave tile 64x32.
  const int lane = tid & 31;
  const int lo = lane & 15;
  const int hi = lane >> 4;
  const int wv = tid >> 5;
  const int wm = wv >> 2;                  // 0..1
  const int wn = wv & 3;                   // 0..3

  v8f acc[4][2] = {};
  float fA[16], fB[16];

  // Prologue: stage K-step 0.
  gload16(aSrc, fA);
  gload16(bSrc, fB);
  lstore16(&lA[0][sOff], fA);
  lstore16(&lB[0][sOff], fB);
  __syncthreads();

  for (int kk = 0; kk < NKSTEP; ++kk) {
    const int cur = kk & 1;

    // Issue next K-step global loads before the WMMAs (latency hiding).
    if (kk + 1 < NKSTEP) {
      const int k0 = (kk + 1) * TILE_K;
      gload16(aSrc + k0, fA);
      gload16(bSrc + k0, fB);
    }

    // A fragments: ISA 16-bit 16x32 layout. Lane lo = row M; hi selects the
    // K sub-chunk: {K 0..7, 16..23} for hi=0, {K 8..15, 24..31} for hi=1.
    v16bf af[4];
    #pragma unroll
    for (int m = 0; m < 4; ++m) {
      const int row = wm * 64 + m * 16 + lo;
      af[m] = frag(&lA[cur][row * TILE_K + hi * 8],
                   &lA[cur][row * TILE_K + 16 + hi * 8]);
    }
    // B fragments: lane lo = column N; hi=0 -> K {0..7, 16..23}, hi=1 -> K {8..15, 24..31}
    // (same K order as A, read along a W row — no transpose needed).
    v16bf bfr[2];
    #pragma unroll
    for (int n = 0; n < 2; ++n) {
      const int col = wn * 32 + n * 16 + lo;
      bfr[n] = frag(&lB[cur][col * TILE_K + hi * 8],
                    &lB[cur][col * TILE_K + 16 + hi * 8]);
    }

    #pragma unroll
    for (int m = 0; m < 4; ++m)
      #pragma unroll
      for (int n = 0; n < 2; ++n)
        acc[m][n] = __builtin_amdgcn_wmma_f32_16x16x32_bf16(
            false, af[m], false, bfr[n], (short)0, acc[m][n], false, false);

    // Convert+stage next buffer (disjoint from the one being read).
    if (kk + 1 < NKSTEP) {
      const int nxt = cur ^ 1;
      lstore16(&lA[nxt][sOff], fA);
      lstore16(&lB[nxt][sOff], fB);
    }
    __syncthreads();
  }

  // C scatter per 16x16 f32 layout: M = v + hi*8, N = lo.
  // Non-temporal: C is written once, never re-read; keep L2 for x/W reuse.
  #pragma unroll
  for (int m = 0; m < 4; ++m) {
    #pragma unroll
    for (int v = 0; v < 8; ++v) {
      const int i = m0 + wm * 64 + m * 16 + v + hi * 8;
      if (i < count) {
        const int tok = idx[i];
        float* o = out + (size_t)tok * DN + n0 + wn * 32 + lo;
        __builtin_nontemporal_store(acc[m][0][v], o);
        __builtin_nontemporal_store(acc[m][1][v], o + 16);
      }
    }
  }
}

extern "C" void kernel_launch(void* const* d_in, const int* in_sizes, int n_in,
                              void* d_out, int out_size, void* d_ws, size_t ws_size,
                              hipStream_t stream) {
  const float*         x    = (const float*)d_in[0];
  const unsigned char* mask = (const unsigned char*)d_in[1];  // jnp.bool_ = 1 byte
  const float*         Wv   = (const float*)d_in[2];
  const float*         Wt   = (const float*)d_in[3];
  float* out = (float*)d_out;

  int* cnts = (int*)d_ws;                       // 2 counters
  int* idxV = (int*)((char*)d_ws + 64);         // 16384 ints
  int* idxT = idxV + MTOK;                      // 16384 ints

  zero_counters<<<1, 32, 0, stream>>>(cnts);
  compact_mask<<<MTOK / 256, 256, 0, stream>>>(mask, cnts, idxV, idxT);

  dim3 grid(DN / TILE_N, MTOK / TILE_M, 2);
  masked_gemm<<<grid, 256, 0, stream>>>(x, Wv, Wt, out, cnts, idxV, idxT);
}
